// CustomLayer_30588757082254
// MI455X (gfx1250) — matrix operand order repaired and checked by reading, not compiled, not run on hardware
//
#include <hip/hip_runtime.h>

#define N_IN   2048
#define N_OUT  2048
#define BATCH  8192

typedef __attribute__((ext_vector_type(16))) __bf16 v16bf;
typedef __attribute__((ext_vector_type(8)))  float  v8f;

// ---------- helpers ----------
static __device__ __forceinline__ unsigned int f2bf_rne(float f) {
  unsigned int u = __float_as_uint(f);
  return (u + 0x7FFFu + ((u >> 16) & 1u)) >> 16;   // round-to-nearest-even bf16
}
static __device__ __forceinline__ unsigned int pk_bf16x2(float lo, float hi) {
  return (f2bf_rne(lo) & 0xFFFFu) | (f2bf_rne(hi) << 16);
}

// ---------- pass 1: global min/max of weight (4M f32) ----------
__global__ __launch_bounds__(256) void minmax_stage1(const float* __restrict__ w,
                                                     float* __restrict__ partials) {
  __shared__ float smin[256], smax[256];
  const int t = threadIdx.x;
  const int gid = blockIdx.x * 256 + t;            // 512 blocks * 256 thr
  float mn = 3.4e38f, mx = -3.4e38f;
  const float4* w4 = (const float4*)w;             // 1,048,576 float4 total
#pragma unroll
  for (int i = 0; i < 8; ++i) {
    float4 v = w4[gid + i * 131072];
    mn = fminf(mn, fminf(fminf(v.x, v.y), fminf(v.z, v.w)));
    mx = fmaxf(mx, fmaxf(fmaxf(v.x, v.y), fmaxf(v.z, v.w)));
  }
  smin[t] = mn; smax[t] = mx;
  __syncthreads();
  for (int s = 128; s > 0; s >>= 1) {
    if (t < s) { smin[t] = fminf(smin[t], smin[t + s]); smax[t] = fmaxf(smax[t], smax[t + s]); }
    __syncthreads();
  }
  if (t == 0) { partials[blockIdx.x] = smin[0]; partials[512 + blockIdx.x] = smax[0]; }
}

__global__ __launch_bounds__(256) void minmax_stage2(const float* __restrict__ partials,
                                                     float* __restrict__ scal) {
  __shared__ float smin[256], smax[256];
  const int t = threadIdx.x;
  smin[t] = fminf(partials[t],       partials[256 + t]);
  smax[t] = fmaxf(partials[512 + t], partials[768 + t]);
  __syncthreads();
  for (int s = 128; s > 0; s >>= 1) {
    if (t < s) { smin[t] = fminf(smin[t], smin[t + s]); smax[t] = fmaxf(smax[t], smax[t + s]); }
    __syncthreads();
  }
  if (t == 0) { scal[0] = smin[0]; scal[1] = smax[0]; }
}

// ---------- pass 2a: fold map + IR-drop + unmap into one bf16 matrix ----------
// Mt[n][k] = S*(G_eff(k,n) - g_min) + w_min, stored N-major (weight layout)
__global__ __launch_bounds__(256) void build_mt(const float* __restrict__ weight,
                                                const float* __restrict__ scal,
                                                unsigned short* __restrict__ Mt) {
  const float g_min = 1.0f / 100000.0f;            // 1/R_HRS
  const float g_max = 1.0f / 1000.0f;              // 1/R_LRS
  const float w_min = scal[0], w_max = scal[1];
  const float w2g = (g_max - g_min) / (w_max - w_min);
  const float g2w = (w_max - w_min) / (g_max - g_min);
  const int idx4 = blockIdx.x * 256 + threadIdx.x; // 4096 blocks -> 1,048,576 float4
  const int n  = idx4 >> 9;                        // out index (crossbar col)
  const int k4 = idx4 & 511;
  float4 wv = ((const float4*)weight)[idx4];
  float wr[4] = {wv.x, wv.y, wv.z, wv.w};
  unsigned short ov[4];
#pragma unroll
  for (int j = 0; j < 4; ++j) {
    int k = k4 * 4 + j;                            // in index (crossbar row)
    float G     = g_min + (wr[j] - w_min) * w2g;
    float R_par = 2.0f * (float)(n + 1) + 2.0f * (float)(N_IN - k);
    float G_eff = 1.0f / (1.0f / G + R_par);
    float m     = (G_eff - g_min) * g2w + w_min;
    ov[j] = (unsigned short)f2bf_rne(m);
  }
  unsigned int lo = (unsigned)ov[0] | ((unsigned)ov[1] << 16);
  unsigned int hi = (unsigned)ov[2] | ((unsigned)ov[3] << 16);
  ((uint2*)Mt)[idx4] = make_uint2(lo, hi);
}

// ---------- pass 2b: pre-convert X to bf16 once (removes VALU from GEMM loop) ----------
__global__ __launch_bounds__(256) void convert_x(const float* __restrict__ x,
                                                 unsigned short* __restrict__ Xbf) {
  const int idx4 = blockIdx.x * 256 + threadIdx.x; // 16384 blocks -> 4,194,304 float4
  float4 v = ((const float4*)x)[idx4];
  uint2 p;
  p.x = pk_bf16x2(v.x, v.y);
  p.y = pk_bf16x2(v.z, v.w);
  ((uint2*)Xbf)[idx4] = p;
}

// ---------- pass 3: C = X * M + bias, bf16 WMMA, f32 accumulate ----------
// BK=64: 32 WMMAs per wave per stage, half the barriers of BK=32.
#define LDA 72   // 64 bf16 + 8 pad -> 144B rows (36 banks): conflict-free frag reads
#define LDB 72

template <bool CONV>
__global__ __launch_bounds__(256) void gemm_ir(const float* __restrict__ X,
                                               const unsigned short* __restrict__ Xbf,
                                               const unsigned short* __restrict__ Mt,
                                               const float* __restrict__ bias,
                                               float* __restrict__ out) {
  __shared__ unsigned short lA[256 * LDA];         // 36 KB, X tile bf16 [M][K]
  __shared__ unsigned short lB[128 * LDB];         // 18 KB, M tile bf16 [N][K]
  const int tid   = threadIdx.x;
  const int lane  = tid & 31;
  const int wave  = tid >> 5;
  const int m0    = blockIdx.y * 256;
  const int n0    = blockIdx.x * 128;
  const int wm0   = (wave >> 1) * 64;              // 4x2 wave grid, 64x64 per wave
  const int wn0   = (wave & 1) * 64;
  const int lhalf = lane >> 4;
  const int l16   = lane & 15;

  v8f acc[4][4];
#pragma unroll
  for (int i = 0; i < 4; ++i)
#pragma unroll
    for (int j = 0; j < 4; ++j)
#pragma unroll
      for (int e = 0; e < 8; ++e) acc[i][j][e] = 0.0f;

  union Frag { uint4 u[2]; v16bf v; };

  for (int kb = 0; kb < N_IN; kb += 64) {
    // ---- stage X tile: 256x64 bf16 ----
    if (CONV) {
#pragma unroll
      for (int i = 0; i < 16; ++i) {
        int f = i * 256 + tid;                     // 4096 float4 in tile
        int row = f >> 4, c4 = f & 15;
        float4 xv = *(const float4*)(X + (size_t)(m0 + row) * N_IN + kb + c4 * 4);
        uint2 p;
        p.x = pk_bf16x2(xv.x, xv.y);
        p.y = pk_bf16x2(xv.z, xv.w);
        *(uint2*)&lA[row * LDA + c4 * 4] = p;
      }
    } else {
#pragma unroll
      for (int i = 0; i < 8; ++i) {
        int u = i * 256 + tid;                     // 2048 uint4 in tile
        int row = u >> 3, seg = u & 7;
        uint4 v = *(const uint4*)(Xbf + (size_t)(m0 + row) * N_IN + kb + seg * 8);
        *(uint4*)&lA[row * LDA + seg * 8] = v;
      }
    }
    // ---- stage Mt tile: 128x64 bf16 (b128 copies) ----
#pragma unroll
    for (int i = 0; i < 4; ++i) {
      int u = i * 256 + tid;                       // 1024 uint4 in tile
      int row = u >> 3, seg = u & 7;
      uint4 v = *(const uint4*)(Mt + (size_t)(n0 + row) * N_IN + kb + seg * 8);
      *(uint4*)&lB[row * LDB + seg * 8] = v;
    }
    __syncthreads();

#pragma unroll
    for (int kk = 0; kk < 64; kk += 32) {
      // B fragments: lane l holds col n, K = {kh*8..+7} and {16+kh*8..+7}
      Frag bfrag[4];
#pragma unroll
      for (int ns = 0; ns < 4; ++ns) {
        const unsigned short* p = &lB[(wn0 + ns * 16 + l16) * LDB + kk + lhalf * 8];
        bfrag[ns].u[0] = *(const uint4*)(p);
        bfrag[ns].u[1] = *(const uint4*)(p + 16);
      }
      // A fragments: lane l holds row (l&15), K = {kh*8..+7} and {16+kh*8..+7}
#pragma unroll
      for (int ms = 0; ms < 4; ++ms) {
        Frag af;
        const unsigned short* p = &lA[(wm0 + ms * 16 + l16) * LDA + kk + lhalf * 8];
        af.u[0] = *(const uint4*)(p);
        af.u[1] = *(const uint4*)(p + 16);
#pragma unroll
        for (int ns = 0; ns < 4; ++ns) {
          acc[ms][ns] = __builtin_amdgcn_wmma_f32_16x16x32_bf16(
              false, af.v, false, bfrag[ns].v, (short)0, acc[ms][ns], false, false);
        }
      }
    }
    __syncthreads();
  }

  // epilogue: C/D layout -> M = v + 8*lhalf, N = l16 ; add bias, store f32
  float bn[4];
#pragma unroll
  for (int ns = 0; ns < 4; ++ns) bn[ns] = bias[n0 + wn0 + ns * 16 + l16];

  const int rbase = m0 + wm0 + 8 * lhalf;
  const int cbase = n0 + wn0 + l16;
#pragma unroll
  for (int ms = 0; ms < 4; ++ms)
#pragma unroll
    for (int ns = 0; ns < 4; ++ns)
#pragma unroll
      for (int v = 0; v < 8; ++v)
        out[(size_t)(rbase + ms * 16 + v) * N_OUT + cbase + ns * 16] =
            acc[ms][ns][v] + bn[ns];
}

// ---------- launch ----------
extern "C" void kernel_launch(void* const* d_in, const int* in_sizes, int n_args,
                              void* d_out, int out_size, void* d_ws, size_t ws_size,
                              hipStream_t stream) {
  (void)in_sizes; (void)n_args; (void)out_size;
  const float* x      = (const float*)d_in[0];
  const float* weight = (const float*)d_in[1];
  const float* bias   = (const float*)d_in[2];
  float* out = (float*)d_out;

  // workspace layout:
  //   [0,16)            scalars (w_min, w_max)
  //   [16, 16+4KB)      reduction partials
  //   [8KB, 8KB+8MB)    Mt bf16 (2048 x 2048)
  //   [8KB+8MB, +32MB)  Xbf bf16 (8192 x 2048)   (optional, if ws is big enough)
  const size_t MT_OFF    = 8192;
  const size_t MT_BYTES  = (size_t)N_OUT * N_IN * 2;
  const size_t XBF_OFF   = MT_OFF + MT_BYTES;
  const size_t XBF_BYTES = (size_t)BATCH * N_IN * 2;

  float* scal            = (float*)d_ws;
  float* partials        = scal + 4;
  unsigned short* Mt     = (unsigned short*)((char*)d_ws + MT_OFF);
  unsigned short* Xbf    = (unsigned short*)((char*)d_ws + XBF_OFF);

  minmax_stage1<<<512, 256, 0, stream>>>(weight, partials);
  minmax_stage2<<<1, 256, 0, stream>>>(partials, scal);
  build_mt<<<4096, 256, 0, stream>>>(weight, scal, Mt);

  dim3 grid(N_OUT / 128, BATCH / 256);
  if (ws_size >= XBF_OFF + XBF_BYTES) {
    convert_x<<<16384, 256, 0, stream>>>(x, Xbf);
    gemm_ir<false><<<grid, 256, 0, stream>>>(x, Xbf, Mt, bias, out);
  } else {
    gemm_ir<true><<<grid, 256, 0, stream>>>(x, Xbf, Mt, bias, out);
  }
}
